// selfatt_64896955843117
// MI455X (gfx1250) — matrix operand order repaired and checked by reading, not compiled, not run on hardware
//
#include <hip/hip_runtime.h>
#include <stdint.h>

#define B_ 4
#define C_ 512
#define L_ 4096
#define DK_ 64
#define CP_ 544      // 513 padded to 17*32
#define KT_ 17       // K tiles of 32 over CP_

typedef __attribute__((ext_vector_type(16))) __bf16 v16bf;
typedef __attribute__((ext_vector_type(8)))  float  v8f;

union bfvec { v16bf v; uint32_t u[8]; };

// native hardware f32->bf16 convert (RNE) instead of manual bit-twiddling
static __device__ inline unsigned short f2bf(float f) {
  __bf16 h = (__bf16)f;
  return __builtin_bit_cast(unsigned short, h);
}

static __device__ inline v8f wmma_bf16(const bfvec& a, const bfvec& b, v8f c) {
  return __builtin_amdgcn_wmma_f32_16x16x32_bf16(
      /*neg_a=*/false, a.v, /*neg_b=*/false, b.v,
      /*c_mod=*/(short)0, c, /*reuse_a=*/false, /*reuse_b=*/false);
}

// ---------------------------------------------------------------------------
// Kernel 1: x0 [B,C,L] f32  ->  xb [B,L,CP] bf16 (transpose, append pos chan,
// zero-pad channels 513..543). LDS-tiled so both global read & write coalesce.
// grid (L/32, CP/32=17, B), block (32,8)
// ---------------------------------------------------------------------------
__global__ void prep_x(const float* __restrict__ x0, unsigned short* __restrict__ xb) {
  int b = blockIdx.z, ch0 = blockIdx.y * 32, l0 = blockIdx.x * 32;
  __shared__ float tile[32][33];
  int tx = threadIdx.x, ty = threadIdx.y;
#pragma unroll
  for (int rr = 0; rr < 4; ++rr) {
    int ch = ch0 + ty + rr * 8;
    int l  = l0 + tx;
    float v;
    if (ch < C_)       v = x0[((size_t)b * C_ + ch) * L_ + l];
    else if (ch == C_) v = (float)l * (2.0f / (float)L_) - 1.0f;
    else               v = 0.0f;
    tile[ty + rr * 8][tx] = v;   // tile[chLocal][lLocal]
  }
  __syncthreads();
#pragma unroll
  for (int rr = 0; rr < 4; ++rr) {
    int l  = l0 + ty + rr * 8;
    int ch = ch0 + tx;
    xb[((size_t)b * L_ + l) * CP_ + ch] = f2bf(tile[tx][ty + rr * 8]);
  }
}

// ---------------------------------------------------------------------------
// Kernel 2: weights f32 [D,513] -> bf16 [D,CP] zero-padded
// ---------------------------------------------------------------------------
__global__ void prep_w(const float* __restrict__ Wq, const float* __restrict__ Wk,
                       const float* __restrict__ Wv,
                       unsigned short* __restrict__ wq, unsigned short* __restrict__ wk,
                       unsigned short* __restrict__ wv) {
  int idx = blockIdx.x * blockDim.x + threadIdx.x;
  const int total = (DK_ + DK_ + C_) * CP_;
  if (idx >= total) return;
  int row = idx / CP_, ch = idx % CP_;
  const float* src; unsigned short* dst; int r;
  if (row < DK_)          { src = Wq; dst = wq; r = row; }
  else if (row < 2 * DK_) { src = Wk; dst = wk; r = row - DK_; }
  else                    { src = Wv; dst = wv; r = row - 2 * DK_; }
  float v = (ch < C_ + 1) ? src[(size_t)r * (C_ + 1) + ch] : 0.0f;
  dst[(size_t)r * CP_ + ch] = f2bf(v);
}

// ---------------------------------------------------------------------------
// Kernel 3: q/k projections via WMMA.
//   qb[b,l,d] = sum_ch xb[l,ch] * Wq[d,ch]   (kb identical with Wk; kb stores
//   k in the same [l,d] layout so the flash kernel's B-fragments read contiguous d-pairs)
// grid (L/16, 2, B), block 128 (4 waves, one d-tile each)
// ---------------------------------------------------------------------------
__global__ __launch_bounds__(128) void qk_proj(const unsigned short* __restrict__ xb,
                                               const unsigned short* __restrict__ wq,
                                               const unsigned short* __restrict__ wk,
                                               unsigned short* __restrict__ qb,
                                               unsigned short* __restrict__ kb) {
  int b = blockIdx.z, l0 = blockIdx.x * 16, which = blockIdx.y;
  int wave = threadIdx.x >> 5, lane = threadIdx.x & 31;
  int half = lane >> 4, mrow = lane & 15;
  int d0 = wave * 16;
  const unsigned short* w = which ? wk : wq;
  unsigned short* dst = which ? kb : qb;
  v8f acc = {};
  for (int kt = 0; kt < KT_; ++kt) {
    int kb0 = kt * 32;
    bfvec a, bm;
#pragma unroll
    for (int v = 0; v < 8; ++v) {
      int ka = (v < 4 ? 0 : 16) + half * 8 + (v & 3) * 2;           // A 16x32 layout
      a.u[v]  = *(const uint32_t*)(xb + ((size_t)b * L_ + l0 + mrow) * CP_ + kb0 + ka);
      int kk = (v < 4 ? 0 : 16) + half * 8 + (v & 3) * 2;           // B 32x16 layout
      bm.u[v] = *(const uint32_t*)(w + (size_t)(d0 + mrow) * CP_ + kb0 + kk);
    }
    acc = wmma_bf16(a, bm, acc);
  }
#pragma unroll
  for (int r = 0; r < 8; ++r) {
    int l = l0 + r + half * 8;                                      // D 16x16 layout
    dst[((size_t)b * L_ + l) * DK_ + d0 + mrow] = f2bf(acc[r]);
  }
}

// ---------------------------------------------------------------------------
// Kernel 4: v projection: vb[b,c,l] = sum_ch Wv[c,ch]*xb[l,ch] + bv[c]  (bf16)
// grid (L/16, C/64, B), block 128 (4 waves, one c-tile each)
// ---------------------------------------------------------------------------
__global__ __launch_bounds__(128) void v_proj(const unsigned short* __restrict__ xb,
                                              const unsigned short* __restrict__ wv,
                                              const float* __restrict__ bv,
                                              unsigned short* __restrict__ vb) {
  int b = blockIdx.z, l0 = blockIdx.x * 16;
  int wave = threadIdx.x >> 5, lane = threadIdx.x & 31;
  int half = lane >> 4, mrow = lane & 15;
  int c0 = (blockIdx.y * 4 + wave) * 16;
  v8f acc = {};
  for (int kt = 0; kt < KT_; ++kt) {
    int kb0 = kt * 32;
    bfvec a, bm;
#pragma unroll
    for (int v = 0; v < 8; ++v) {
      int ka = (v < 4 ? 0 : 16) + half * 8 + (v & 3) * 2;
      a.u[v]  = *(const uint32_t*)(wv + (size_t)(c0 + mrow) * CP_ + kb0 + ka);
      int kk = (v < 4 ? 0 : 16) + half * 8 + (v & 3) * 2;
      bm.u[v] = *(const uint32_t*)(xb + ((size_t)b * L_ + l0 + mrow) * CP_ + kb0 + kk);
    }
    acc = wmma_bf16(a, bm, acc);
  }
#pragma unroll
  for (int r = 0; r < 8; ++r) {
    int c = c0 + r + half * 8;
    vb[((size_t)b * C_ + c) * L_ + l0 + mrow] = f2bf(acc[r] + bv[c]);
  }
}

// ---------------------------------------------------------------------------
// Kernel 5: flash attention + epilogue, 32 query rows per WG.
// 10 waves:
//   waves 8,9: independent softmax producers, one 16-row i-tile each.
//     S = q·k^T (bf16 WMMA, f32 acc), online softmax (shfl_xor row
//     reductions), write P tile (bf16 [16 i][32 j]) + per-row scale into
//     double-buffered LDS.
//   waves 0-7: each owns 64 output channels; per 32-j step: one vb A-fragment
//     feeds TWO WMMAs (both i-tiles), doubling math per byte of vb traffic.
// Epilogue: out = gamma*acc/l + x0 (coalesced along i).
// grid (L/32, B), block 320
// ---------------------------------------------------------------------------
__global__ __launch_bounds__(320) void flash(const unsigned short* __restrict__ qb,
                                             const unsigned short* __restrict__ kbuf,
                                             const unsigned short* __restrict__ vb,
                                             const float* __restrict__ x0,
                                             const float* __restrict__ gamma,
                                             float* __restrict__ out) {
  int b = blockIdx.y, i0 = blockIdx.x * 32;
  int wave = threadIdx.x >> 5, lane = threadIdx.x & 31;
  int half = lane >> 4, mrow = lane & 15;

  __shared__ __align__(4) unsigned short P[2][2][16 * 32];  // [buf][itile][m*32+j]
  __shared__ float Sc[2][2][16];
  __shared__ float Linv[2][16];

  v8f acc[4][2];
#pragma unroll
  for (int t = 0; t < 4; ++t) { acc[t][0] = v8f{}; acc[t][1] = v8f{}; }

  bfvec qA[2];
  float m_run[8], l_run[8];
  const int it = wave - 8;   // producer's i-tile (0/1), valid when wave >= 8
  if (wave >= 8) {
#pragma unroll
    for (int kt = 0; kt < 2; ++kt)
#pragma unroll
      for (int v = 0; v < 8; ++v) {
        int ka = kt * 32 + (v < 4 ? 0 : 16) + half * 8 + (v & 3) * 2;
        qA[kt].u[v] = *(const uint32_t*)(qb + ((size_t)b * L_ + i0 + it * 16 + mrow) * DK_ + ka);
      }
#pragma unroll
    for (int r = 0; r < 8; ++r) { m_run[r] = -3.0e38f; l_run[r] = 0.0f; }
  }

  for (int j0 = 0; j0 < L_; j0 += 32) {
    int buf = (j0 >> 5) & 1;
    if (wave >= 8) {
      v8f st[2];
#pragma unroll
      for (int jt = 0; jt < 2; ++jt) {
        v8f s = {};
#pragma unroll
        for (int kt = 0; kt < 2; ++kt) {
          bfvec kB;
#pragma unroll
          for (int v = 0; v < 8; ++v) {
            int d = kt * 32 + (v < 4 ? 0 : 16) + half * 8 + (v & 3) * 2;
            kB.u[v] = *(const uint32_t*)(kbuf + ((size_t)b * L_ + j0 + jt * 16 + mrow) * DK_ + d);
          }
          s = wmma_bf16(qA[kt], kB, s);
        }
        st[jt] = s;
      }
      // online softmax update for the 8 rows owned by this (r, half) slot
#pragma unroll
      for (int r = 0; r < 8; ++r) {
        float vm = fmaxf(st[0][r], st[1][r]);
        vm = fmaxf(vm, __shfl_xor(vm, 1, 32));
        vm = fmaxf(vm, __shfl_xor(vm, 2, 32));
        vm = fmaxf(vm, __shfl_xor(vm, 4, 32));
        vm = fmaxf(vm, __shfl_xor(vm, 8, 32));
        float mnew  = fmaxf(m_run[r], vm);
        float scale = __expf(m_run[r] - mnew);
        float p0 = __expf(st[0][r] - mnew);
        float p1 = __expf(st[1][r] - mnew);
        float rs = p0 + p1;
        rs += __shfl_xor(rs, 1, 32);
        rs += __shfl_xor(rs, 2, 32);
        rs += __shfl_xor(rs, 4, 32);
        rs += __shfl_xor(rs, 8, 32);
        l_run[r] = l_run[r] * scale + rs;
        m_run[r] = mnew;
        int m = r + half * 8;
        P[buf][it][m * 32 + mrow]      = f2bf(p0);
        P[buf][it][m * 32 + 16 + mrow] = f2bf(p1);
        if (mrow == 0) Sc[buf][it][m] = scale;
      }
    }
    __syncthreads();
    if (wave < 8) {
      float sc0 = Sc[buf][0][mrow];        // scale per attention row (= D col n)
      float sc1 = Sc[buf][1][mrow];
      bfvec pB0, pB1;
#pragma unroll
      for (int v = 0; v < 8; ++v) {
        int k = (v < 4 ? 0 : 16) + half * 8 + (v & 3) * 2;
        pB0.u[v] = *(const uint32_t*)&P[buf][0][mrow * 32 + k];
        pB1.u[v] = *(const uint32_t*)&P[buf][1][mrow * 32 + k];
      }
#pragma unroll
      for (int t = 0; t < 4; ++t) {
        int c0 = (wave * 4 + t) * 16;
        bfvec aV;
#pragma unroll
        for (int v = 0; v < 8; ++v) {
          int k = (v < 4 ? 0 : 16) + half * 8 + (v & 3) * 2;
          aV.u[v] = *(const uint32_t*)(vb + ((size_t)b * C_ + c0 + mrow) * L_ + j0 + k);
        }
        v8f c0v = acc[t][0];
        v8f c1v = acc[t][1];
#pragma unroll
        for (int r = 0; r < 8; ++r) { c0v[r] *= sc0; c1v[r] *= sc1; }
        acc[t][0] = wmma_bf16(aV, pB0, c0v);
        acc[t][1] = wmma_bf16(aV, pB1, c1v);
      }
    }
  }

  if (wave >= 8) {
    float g = gamma[0];
    if (mrow == 0) {
#pragma unroll
      for (int r = 0; r < 8; ++r) Linv[it][r + half * 8] = g / l_run[r];
    }
  }
  __syncthreads();
  if (wave < 8) {
    float li0 = Linv[0][mrow];
    float li1 = Linv[1][mrow];
#pragma unroll
    for (int t = 0; t < 4; ++t) {
      int c0 = (wave * 4 + t) * 16;
#pragma unroll
      for (int r = 0; r < 8; ++r) {
        int c = c0 + r + half * 8;
        size_t idx0 = ((size_t)b * C_ + c) * L_ + i0 + mrow;
        out[idx0]      = acc[t][0][r] * li0 + x0[idx0];
        out[idx0 + 16] = acc[t][1][r] * li1 + x0[idx0 + 16];
      }
    }
  }
}

// ---------------------------------------------------------------------------
extern "C" void kernel_launch(void* const* d_in, const int* in_sizes, int n_in,
                              void* d_out, int out_size, void* d_ws, size_t ws_size,
                              hipStream_t stream) {
  (void)in_sizes; (void)n_in; (void)out_size; (void)ws_size;
  const float* x0    = (const float*)d_in[0];
  const float* Wq    = (const float*)d_in[1];
  const float* Wk    = (const float*)d_in[2];
  const float* Wv    = (const float*)d_in[3];
  const float* bv    = (const float*)d_in[4];
  const float* gamma = (const float*)d_in[5];
  float* out = (float*)d_out;

  // workspace layout (bytes)
  char* ws = (char*)d_ws;
  size_t o = 0;
  unsigned short* xb = (unsigned short*)(ws + o); o += (size_t)B_ * L_ * CP_ * 2;   // 17.8 MB
  unsigned short* wq = (unsigned short*)(ws + o); o += (size_t)DK_ * CP_ * 2;
  unsigned short* wk = (unsigned short*)(ws + o); o += (size_t)DK_ * CP_ * 2;
  unsigned short* wv = (unsigned short*)(ws + o); o += (size_t)C_ * CP_ * 2;
  unsigned short* qb = (unsigned short*)(ws + o); o += (size_t)B_ * L_ * DK_ * 2;   // 2 MB
  unsigned short* kb = (unsigned short*)(ws + o); o += (size_t)B_ * L_ * DK_ * 2;   // 2 MB
  unsigned short* vb = (unsigned short*)(ws + o); o += (size_t)B_ * C_ * L_ * 2;    // 16.8 MB

  prep_x<<<dim3(L_ / 32, CP_ / 32, B_), dim3(32, 8), 0, stream>>>(x0, xb);
  {
    int total = (DK_ + DK_ + C_) * CP_;
    prep_w<<<(total + 255) / 256, 256, 0, stream>>>(Wq, Wk, Wv, wq, wk, wv);
  }
  qk_proj<<<dim3(L_ / 16, 2, B_), 128, 0, stream>>>(xb, wq, wk, qb, kb);
  v_proj<<<dim3(L_ / 16, C_ / 64, B_), 128, 0, stream>>>(xb, wv, bv, vb);
  flash<<<dim3(L_ / 32, B_), 320, 0, stream>>>(qb, kb, vb, x0, gamma, out);
}
